// GraphGPSBackbone_44805098832147
// MI455X (gfx1250) — matrix operand order repaired and checked by reading, not compiled, not run on hardware
//
#include <hip/hip_runtime.h>

// GraphGPS backbone on gfx1250 (MI455X): WMMA f16->f32 GEMMs + flash attention
// + fused GINE edge scatter. Wave32. Async global->LDS DMA for GEMM A tiles.

typedef _Float16 v16h __attribute__((ext_vector_type(16)));
typedef _Float16 v8h  __attribute__((ext_vector_type(8)));
typedef float    v8f  __attribute__((ext_vector_type(8)));
typedef float    v4f  __attribute__((ext_vector_type(4)));

#define NTOT  32768
#define DD    256
#define DIN_  64
#define NE    1048576
#define NB    128
#define NH    8
#define NL    4
#define BN_SCALE_F 0.9999950000374997f   // 1/sqrt(1+1e-5)

__device__ __forceinline__ v16h cat16(v8h lo, v8h hi) {
  return __builtin_shufflevector(lo, hi, 0,1,2,3,4,5,6,7,8,9,10,11,12,13,14,15);
}

__device__ __forceinline__ v16h cvt16(v4f a0, v4f a1, v4f a2, v4f a3) {
  v16h r;
  r[0]=(_Float16)a0.x;  r[1]=(_Float16)a0.y;  r[2]=(_Float16)a0.z;  r[3]=(_Float16)a0.w;
  r[4]=(_Float16)a1.x;  r[5]=(_Float16)a1.y;  r[6]=(_Float16)a1.z;  r[7]=(_Float16)a1.w;
  r[8]=(_Float16)a2.x;  r[9]=(_Float16)a2.y;  r[10]=(_Float16)a2.z; r[11]=(_Float16)a2.w;
  r[12]=(_Float16)a3.x; r[13]=(_Float16)a3.y; r[14]=(_Float16)a3.z; r[15]=(_Float16)a3.w;
  return r;
}

__device__ __forceinline__ void async_wait0() {
#if defined(__has_builtin) && __has_builtin(__builtin_amdgcn_s_wait_asynccnt)
  __builtin_amdgcn_s_wait_asynccnt(0);
#else
  asm volatile("s_wait_asynccnt 0x0" ::: "memory");
#endif
}

// issue one per-lane 16B async DMA: global[saddr + voff] -> LDS[ldsoff]
__device__ __forceinline__ void async_copy_b128(unsigned ldsoff, unsigned voff,
                                                unsigned long long sbase) {
  asm volatile("global_load_async_to_lds_b128 %0, %1, %2"
               :: "v"(ldsoff), "v"(voff), "s"(sbase) : "memory");
}

// ---------------------------------------------------------------------------
// Generic WMMA GEMM: C[M,N] = epi(A[M,K] @ B[K,N])
// block = 256 threads (8 waves), tile 128x64, K-step 32.
// A tile: async DMA (f32) -> LDS, cvt at fragment build.
// B tile: VGPR transpose -> f16 LDS (column-major for B fragments).
// epilogue: +bias, [relu], [+res], [BN: *g*s + b], [+res2]
// ---------------------------------------------------------------------------
template<bool RELU, bool HAS_RES, bool HAS_BN, bool HAS_RES2>
__global__ __launch_bounds__(256) void gemm_wmma_k(
    const float* __restrict__ A, const float* __restrict__ Bw,
    const float* __restrict__ bias, const float* __restrict__ res,
    const float* __restrict__ bng, const float* __restrict__ bnb,
    const float* __restrict__ res2, float* __restrict__ C,
    int M, int N, int K)
{
  __shared__ alignas(16) float    Asf[128 * 40];   // f32 A tile, pad 40
  __shared__ alignas(16) _Float16 Bt[64 * 40];     // f16 B^T tile, pad 40

  const int t    = threadIdx.x;
  const int lane = t & 31, w = t >> 5;
  const int wm   = w & 3,  wn = w >> 2;
  const int g    = lane >> 4, l16 = lane & 15;
  const int mBase = blockIdx.x * 128;
  const int nBase = blockIdx.y * 64;

  v8f acc[2][2] = {};

  const int arow = t >> 1;               // A DMA: 0..127
  const int acol = (t & 1) * 16;         // 0 or 16 (floats)
  const int bk = t >> 3;                 // B stage: k 0..31, 8 cols
  const int bn = (t & 7) * 8;
  const unsigned long long abase = (unsigned long long)(uintptr_t)A;

  for (int k0 = 0; k0 < K; k0 += 32) {
    const float* bp = Bw + (size_t)(k0 + bk) * N + nBase + bn;
    v4f bv0 = *(const v4f*)bp;
    v4f bv1 = *(const v4f*)(bp + 4);
    if (k0 + 32 < K)
      __builtin_prefetch(bp + (size_t)32 * N, 0, 1);   // next B k-tile

    __syncthreads();   // previous tile fully consumed

    // A tile via async DMA: 128x32 f32, 16B per lane per instr, 4 per thread
#pragma unroll
    for (int q = 0; q < 4; ++q) {
      unsigned ldsoff = (unsigned)(uintptr_t)&Asf[arow * 40 + acol + q * 4];
      unsigned voff = (unsigned)((((unsigned)(mBase + arow)) * (unsigned)K +
                                  (unsigned)(k0 + acol + q * 4)) * 4u);
      async_copy_b128(ldsoff, voff, abase);
    }

    // B tile transpose -> f16 LDS (overlaps with DMA)
    Bt[(bn + 0) * 40 + bk] = (_Float16)bv0.x;
    Bt[(bn + 1) * 40 + bk] = (_Float16)bv0.y;
    Bt[(bn + 2) * 40 + bk] = (_Float16)bv0.z;
    Bt[(bn + 3) * 40 + bk] = (_Float16)bv0.w;
    Bt[(bn + 4) * 40 + bk] = (_Float16)bv1.x;
    Bt[(bn + 5) * 40 + bk] = (_Float16)bv1.y;
    Bt[(bn + 6) * 40 + bk] = (_Float16)bv1.z;
    Bt[(bn + 7) * 40 + bk] = (_Float16)bv1.w;

    async_wait0();       // this wave's DMAs landed in LDS
    __syncthreads();     // all waves' DMAs + B stores visible

    v16h af[2], bf[2];
#pragma unroll
    for (int ti = 0; ti < 2; ++ti) {   // A frag: row l16, K = {8g..}{16+8g..}
      const float* rp = &Asf[(wm * 32 + ti * 16 + l16) * 40];
      v4f a0 = *(const v4f*)(rp + 8 * g);
      v4f a1 = *(const v4f*)(rp + 8 * g + 4);
      v4f a2 = *(const v4f*)(rp + 16 + 8 * g);
      v4f a3 = *(const v4f*)(rp + 16 + 8 * g + 4);
      af[ti] = cvt16(a0, a1, a2, a3);
    }
#pragma unroll
    for (int tj = 0; tj < 2; ++tj) {   // B frag: col l16, K = {8g..}{16+8g..}
      const _Float16* cp = &Bt[(wn * 32 + tj * 16 + l16) * 40 + 8 * g];
      bf[tj] = cat16(*(const v8h*)cp, *(const v8h*)(cp + 16));
    }
#pragma unroll
    for (int ti = 0; ti < 2; ++ti)
#pragma unroll
      for (int tj = 0; tj < 2; ++tj)
        acc[ti][tj] = __builtin_amdgcn_wmma_f32_16x16x32_f16(
            false, af[ti], false, bf[tj], (short)0, acc[ti][tj], false, false);
  }

#pragma unroll
  for (int ti = 0; ti < 2; ++ti) {
#pragma unroll
    for (int tj = 0; tj < 2; ++tj) {
      const int col = nBase + wn * 32 + tj * 16 + l16;
      const float bcol = bias[col];
      float gs = 0.f, bb = 0.f;
      if constexpr (HAS_BN) { gs = bng[col] * BN_SCALE_F; bb = bnb[col]; }
#pragma unroll
      for (int j = 0; j < 8; ++j) {
        const int row = mBase + wm * 32 + ti * 16 + 8 * g + j;
        const size_t idx = (size_t)row * N + col;
        float v = acc[ti][tj][j] + bcol;
        if constexpr (RELU)     v = fmaxf(v, 0.f);
        if constexpr (HAS_RES)  v += res[idx];
        if constexpr (HAS_BN)   v = v * gs + bb;
        if constexpr (HAS_RES2) v += res2[idx];
        C[idx] = v;
      }
    }
  }
}

// ---------------------------------------------------------------------------
// Per-(graph, head) flash attention: N=256 keys, DH=32, online softmax.
// block = 256 threads (8 waves), each wave owns 16 query rows, 2 iters.
// ---------------------------------------------------------------------------
__global__ __launch_bounds__(256) void attn_wmma_k(const float* __restrict__ qkv,
                                                   float* __restrict__ O)
{
  __shared__ alignas(16) _Float16 Ks[256 * 40];       // K rows (d contiguous)
  __shared__ alignas(16) _Float16 Vst[32 * 264];      // V transposed [d][node]
  __shared__ alignas(16) _Float16 Ps[8 * 16 * 40];    // per-wave P transpose

  const int b = blockIdx.x, head = blockIdx.y;
  const int t = threadIdx.x;
  {
    const float* kr = qkv + (size_t)(b * 256 + t) * 768 + 256 + head * 32;
    const float* vr = kr + 256;
#pragma unroll
    for (int d = 0; d < 32; d += 4) {
      v4f kv = *(const v4f*)(kr + d);
      Ks[t * 40 + d + 0] = (_Float16)kv.x;
      Ks[t * 40 + d + 1] = (_Float16)kv.y;
      Ks[t * 40 + d + 2] = (_Float16)kv.z;
      Ks[t * 40 + d + 3] = (_Float16)kv.w;
      v4f vv = *(const v4f*)(vr + d);
      Vst[(d + 0) * 264 + t] = (_Float16)vv.x;
      Vst[(d + 1) * 264 + t] = (_Float16)vv.y;
      Vst[(d + 2) * 264 + t] = (_Float16)vv.z;
      Vst[(d + 3) * 264 + t] = (_Float16)vv.w;
    }
  }
  __syncthreads();

  const int lane = t & 31, w = t >> 5;
  const int g = lane >> 4, l16 = lane & 15;
  const float scale = 0.1767766952966369f;   // 1/sqrt(32)
  _Float16* PsW = &Ps[w * 16 * 40];

  for (int iter = 0; iter < 2; ++iter) {
    const int q0 = iter * 128 + w * 16;
    v16h qa;   // Q A-fragment straight from global
    {
      const float* qr = qkv + (size_t)(b * 256 + q0 + l16) * 768 + head * 32;
      v4f x0 = *(const v4f*)(qr + 8 * g);
      v4f x1 = *(const v4f*)(qr + 8 * g + 4);
      v4f x2 = *(const v4f*)(qr + 16 + 8 * g);
      v4f x3 = *(const v4f*)(qr + 16 + 8 * g + 4);
      qa = cvt16(x0, x1, x2, x3);
    }

    float mrow[8], lrow[8];
    v8f oacc[2] = {};
#pragma unroll
    for (int j = 0; j < 8; ++j) { mrow[j] = -1e30f; lrow[j] = 0.f; }

    for (int kb = 0; kb < 8; ++kb) {         // 32 keys per block
      v8f s[2];
#pragma unroll
      for (int u = 0; u < 2; ++u) {
        const _Float16* kp = &Ks[(kb * 32 + u * 16 + l16) * 40 + 8 * g];
        v16h kf = cat16(*(const v8h*)kp, *(const v8h*)(kp + 16));
        v8f zc = {};
        s[u] = __builtin_amdgcn_wmma_f32_16x16x32_f16(
            false, qa, false, kf, (short)0, zc, false, false);
      }
#pragma unroll
      for (int j = 0; j < 8; ++j) {          // online softmax, row = j + 8g
        float tm = fmaxf(s[0][j], s[1][j]) * scale;
        tm = fmaxf(tm, __shfl_xor(tm, 1, 32));
        tm = fmaxf(tm, __shfl_xor(tm, 2, 32));
        tm = fmaxf(tm, __shfl_xor(tm, 4, 32));
        tm = fmaxf(tm, __shfl_xor(tm, 8, 32));
        const float mnew = fmaxf(mrow[j], tm);
        const float al = __expf(mrow[j] - mnew);
        const float p0 = __expf(s[0][j] * scale - mnew);
        const float p1 = __expf(s[1][j] * scale - mnew);
        float ps = p0 + p1;
        ps += __shfl_xor(ps, 1, 32);
        ps += __shfl_xor(ps, 2, 32);
        ps += __shfl_xor(ps, 4, 32);
        ps += __shfl_xor(ps, 8, 32);
        lrow[j] = lrow[j] * al + ps;
        mrow[j] = mnew;
        const int pr = (j + 8 * g) * 40 + l16;
        PsW[pr]      = (_Float16)p0;
        PsW[pr + 16] = (_Float16)p1;
        oacc[0][j] *= al;
        oacc[1][j] *= al;
      }
      v16h pa;   // P re-enters as A fragment via LDS transpose
      {
        const _Float16* pp = &PsW[l16 * 40];
        pa = cat16(*(const v8h*)(pp + 8 * g), *(const v8h*)(pp + 16 + 8 * g));
      }
#pragma unroll
      for (int dt = 0; dt < 2; ++dt) {
        const _Float16* vp = &Vst[(dt * 16 + l16) * 264 + kb * 32 + 8 * g];
        v16h vf = cat16(*(const v8h*)vp, *(const v8h*)(vp + 16));
        oacc[dt] = __builtin_amdgcn_wmma_f32_16x16x32_f16(
            false, pa, false, vf, (short)0, oacc[dt], false, false);
      }
    }
#pragma unroll
    for (int dt = 0; dt < 2; ++dt)
#pragma unroll
      for (int j = 0; j < 8; ++j) {
        const int node = b * 256 + q0 + 8 * g + j;
        const int col  = head * 32 + dt * 16 + l16;
        O[(size_t)node * 256 + col] = oacc[dt][j] / lrow[j];
      }
  }
}

// ---------------------------------------------------------------------------
// z = h (float4 copy), then scatter: z[dst] += relu(h[src] + ea*ew + eb)
// ---------------------------------------------------------------------------
__global__ __launch_bounds__(256) void copy_f32x4_k(float* __restrict__ dst,
                                                    const float* __restrict__ src,
                                                    int n4)
{
  int i = blockIdx.x * 256 + threadIdx.x;
  if (i < n4) ((v4f*)dst)[i] = ((const v4f*)src)[i];
}

__global__ __launch_bounds__(256) void gine_scatter_k(
    const float* __restrict__ h, float* __restrict__ z,
    const int* __restrict__ src, const int* __restrict__ dst,
    const float* __restrict__ ea, const float* __restrict__ ew,
    const float* __restrict__ eb, int nedges)
{
  __shared__ float sew[DD], seb[DD];
  const int t = threadIdx.x;
  sew[t] = ew[t];
  seb[t] = eb[t];
  __syncthreads();
  const int lane = t & 31;
  const int wave = blockIdx.x * 8 + (t >> 5);
  const int nwaves = gridDim.x * 8;
  for (int e = wave; e < nedges; e += nwaves) {
    const int s = src[e], d = dst[e];
    const float a = ea[e];
    const int en = e + nwaves;
    if (en < nedges) {                       // prefetch next source row
      const int sn = src[en];
      __builtin_prefetch(h + (size_t)sn * DD + lane * 8, 0, 1);
    }
#pragma unroll
    for (int r = 0; r < 2; ++r) {
      const int c = lane * 4 + r * 128;
      v4f hv = *(const v4f*)(h + (size_t)s * DD + c);
      float m0 = fmaxf(hv.x + a * sew[c + 0] + seb[c + 0], 0.f);
      float m1 = fmaxf(hv.y + a * sew[c + 1] + seb[c + 1], 0.f);
      float m2 = fmaxf(hv.z + a * sew[c + 2] + seb[c + 2], 0.f);
      float m3 = fmaxf(hv.w + a * sew[c + 3] + seb[c + 3], 0.f);
      float* zp = z + (size_t)d * DD + c;
      unsafeAtomicAdd(zp + 0, m0);
      unsafeAtomicAdd(zp + 1, m1);
      unsafeAtomicAdd(zp + 2, m2);
      unsafeAtomicAdd(zp + 3, m3);
    }
  }
}

// ---------------------------------------------------------------------------
extern "C" void kernel_launch(void* const* d_in, const int* in_sizes, int n_in,
                              void* d_out, int out_size, void* d_ws, size_t ws_size,
                              hipStream_t stream)
{
  (void)in_sizes; (void)n_in; (void)out_size; (void)ws_size;
  const float* x          = (const float*)d_in[0];
  const float* edge_attr  = (const float*)d_in[1];
  const float* node_w     = (const float*)d_in[2];
  const float* node_b     = (const float*)d_in[3];
  const float* edge_w     = (const float*)d_in[4];
  const float* edge_b     = (const float*)d_in[5];
  const float* gine_w1    = (const float*)d_in[6];
  const float* gine_b1    = (const float*)d_in[7];
  const float* gine_w2    = (const float*)d_in[8];
  const float* gine_b2    = (const float*)d_in[9];
  const float* attn_in_w  = (const float*)d_in[10];
  const float* attn_in_b  = (const float*)d_in[11];
  const float* attn_out_w = (const float*)d_in[12];
  const float* attn_out_b = (const float*)d_in[13];
  const float* mlp_w1     = (const float*)d_in[14];
  const float* mlp_b1     = (const float*)d_in[15];
  const float* mlp_w2     = (const float*)d_in[16];
  const float* mlp_b2     = (const float*)d_in[17];
  const float* bn1_g = (const float*)d_in[18];
  const float* bn1_b = (const float*)d_in[19];
  const float* bn2_g = (const float*)d_in[20];
  const float* bn2_b = (const float*)d_in[21];
  const float* bn3_g = (const float*)d_in[22];
  const float* bn3_b = (const float*)d_in[23];
  const int* edge_index = (const int*)d_in[24];
  const int* srcp = edge_index;
  const int* dstp = edge_index + NE;

  float* ws = (float*)d_ws;
  const size_t nd = (size_t)NTOT * DD;
  float* h    = ws;            // node features (persists across layer)
  float* h1   = ws + nd;       // local branch
  float* outb = ws + 2 * nd;   // combined branch
  float* zob  = ws + 3 * nd;   // shared scratch: GINE z, then attn o
  float* big  = ws + 4 * nd;   // NTOT x 768: GINE hidden / qkv / mlp hidden

  dim3 blk(256);
  dim3 gD(NTOT / 128, DD / 64);
  dim3 gQ(NTOT / 128, (3 * DD) / 64);
  dim3 gM(NTOT / 128, (2 * DD) / 64);

  // h = x @ node_w + node_b
  gemm_wmma_k<false, false, false, false><<<gD, blk, 0, stream>>>(
      x, node_w, node_b, nullptr, nullptr, nullptr, nullptr, h, NTOT, DD, DIN_);

  for (int l = 0; l < NL; ++l) {
    // z = h + segment_sum(relu(h[src] + ea*ew + eb), dst)
    copy_f32x4_k<<<dim3((NTOT * DD / 4) / 256), blk, 0, stream>>>(zob, h, NTOT * DD / 4);
    gine_scatter_k<<<dim3(4096), blk, 0, stream>>>(h, zob, srcp, dstp, edge_attr,
                                                   edge_w, edge_b, NE);
    // big = relu(z @ W1 + b1)
    gemm_wmma_k<true, false, false, false><<<gD, blk, 0, stream>>>(
        zob, gine_w1 + (size_t)l * DD * DD, gine_b1 + l * DD,
        nullptr, nullptr, nullptr, nullptr, big, NTOT, DD, DD);
    // h1 = bn1(big @ W2 + b2 + h)
    gemm_wmma_k<false, true, true, false><<<gD, blk, 0, stream>>>(
        big, gine_w2 + (size_t)l * DD * DD, gine_b2 + l * DD,
        h, bn1_g + l * DD, bn1_b + l * DD, nullptr, h1, NTOT, DD, DD);
    // big = h @ attn_in_w + b  (qkv)
    gemm_wmma_k<false, false, false, false><<<gQ, blk, 0, stream>>>(
        h, attn_in_w + (size_t)l * DD * 3 * DD, attn_in_b + l * 3 * DD,
        nullptr, nullptr, nullptr, nullptr, big, NTOT, 3 * DD, DD);
    // zob = attention(big)
    attn_wmma_k<<<dim3(NB, NH), blk, 0, stream>>>(big, zob);
    // outb = bn2(zob @ Wo + bo + h) + h1
    gemm_wmma_k<false, true, true, true><<<gD, blk, 0, stream>>>(
        zob, attn_out_w + (size_t)l * DD * DD, attn_out_b + l * DD,
        h, bn2_g + l * DD, bn2_b + l * DD, h1, outb, NTOT, DD, DD);
    // big = relu(outb @ mlp_w1 + b1)
    gemm_wmma_k<true, false, false, false><<<gM, blk, 0, stream>>>(
        outb, mlp_w1 + (size_t)l * DD * 2 * DD, mlp_b1 + l * 2 * DD,
        nullptr, nullptr, nullptr, nullptr, big, NTOT, 2 * DD, DD);
    // h = bn3(big @ mlp_w2 + b2 + outb)   (last layer -> d_out)
    float* hout = (l == NL - 1) ? (float*)d_out : h;
    gemm_wmma_k<false, true, true, false><<<gD, blk, 0, stream>>>(
        big, mlp_w2 + (size_t)l * 2 * DD * DD, mlp_b2 + l * DD,
        outb, bn3_g + l * DD, bn3_b + l * DD, nullptr, hout, NTOT, DD, 2 * DD);
  }
}
